// ViPBlock_28260884808159
// MI455X (gfx1250) — matrix operand order repaired and checked by reading, not compiled, not run on hardware
//
#include <hip/hip_runtime.h>

typedef __attribute__((ext_vector_type(16))) _Float16 v16h;
typedef __attribute__((ext_vector_type(8)))  _Float16 v8h;
typedef __attribute__((ext_vector_type(8)))  float    v8f;

#define T_TOK  100352   // B*P*K = 32*64*49
#define TB_ENC 3136     // P*K per batch
#define NWIN   2048     // B*P windows
#define KW     49
#define DIMC   128

// ---------------------------------------------------------------------------
// WMMA helpers (gfx1250 wave32, 16x16x32 f16 -> f32)
// ---------------------------------------------------------------------------
__device__ __forceinline__ v8f wmma16(const v16h& a, const v16h& b, const v8f& c) {
  return __builtin_amdgcn_wmma_f32_16x16x32_f16(false, a, false, b, (short)0, c, false, false);
}

__device__ __forceinline__ v8f vzero8() {
  v8f z;
#pragma unroll
  for (int i = 0; i < 8; ++i) z[i] = 0.f;
  return z;
}

// A fragment: 16x32 tile of row-major f16 X (pre-offset to [m0][k0]), leading dim ld.
// Layout: row=lane&15, elems j -> K = hb + j (contiguous 16 halves); hb=(lane>>4)*16
__device__ __forceinline__ v16h load_a(const _Float16* __restrict__ X, int ld, int lane) {
  v16h a;
  const int row = lane & 15;
  const int hb  = (lane >> 4) * 16;
  const _Float16* p = X + row * ld;
#pragma unroll
  for (int j = 0; j < 16; ++j) a[j] = p[hb + j];
  return a;
}

// B fragment from "weight-style" W[N,K] row-major (pre-offset to [n0][k0]), leading dim ld.
// Lane holds column n=n0+(lane&15); elems j -> K = (lane>>4)*16 + j  (contiguous 16 halves)
__device__ __forceinline__ v16h load_b(const _Float16* __restrict__ W, int ld, int lane) {
  v16h b;
  const int col = lane & 15;
  const int hb  = (lane >> 4) * 16;
  const _Float16* p = W + col * ld + hb;
#pragma unroll
  for (int j = 0; j < 16; ++j) b[j] = p[j];
  return b;
}

// B fragment from row-major B[K,N] (pre-offset to [k0][n0]), leading dim ld (strided reads).
__device__ __forceinline__ v16h load_bT(const _Float16* __restrict__ Bm, int ld, int lane) {
  v16h b;
  const int col = lane & 15;
  const int hb  = (lane >> 4) * 16;
#pragma unroll
  for (int j = 0; j < 16; ++j) b[j] = Bm[(hb + j) * ld + col];
  return b;
}

__device__ __forceinline__ float redmax16(float v) {
  v = fmaxf(v, __shfl_xor(v, 1, 32));
  v = fmaxf(v, __shfl_xor(v, 2, 32));
  v = fmaxf(v, __shfl_xor(v, 4, 32));
  v = fmaxf(v, __shfl_xor(v, 8, 32));
  return v;
}
__device__ __forceinline__ float redsum16(float v) {
  v += __shfl_xor(v, 1, 32);
  v += __shfl_xor(v, 2, 32);
  v += __shfl_xor(v, 4, 32);
  v += __shfl_xor(v, 8, 32);
  return v;
}

__device__ __forceinline__ float gelu_exact(float x) {
  return 0.5f * x * (1.0f + erff(x * 0.70710678118654752f));
}

// Wave-cooperative LayerNorm of 16 tokens x 128ch -> f16 LDS tile [16][128].
// lane&15 = token, lane>>4 = which 64-channel half; values cached in registers
// (single global read). Optional full-dim pos add.
__device__ __forceinline__ void wave_ln16(const float* __restrict__ X,
                                          const float* __restrict__ pos,
                                          const float* __restrict__ g,
                                          const float* __restrict__ bt,
                                          _Float16* __restrict__ sx, int lane) {
  const int t    = lane & 15;
  const int half = lane >> 4;
  const float* xr = X + t * DIMC + half * 64;
  const float* pr = pos ? (pos + t * DIMC + half * 64) : nullptr;
  float loc[64];
  float s = 0.f, s2 = 0.f;
#pragma unroll
  for (int i = 0; i < 64; ++i) {
    float x = xr[i];
    if (pr) x += pr[i];
    loc[i] = x;
    s += x; s2 += x * x;
  }
  s  += __shfl_xor(s, 16, 32);
  s2 += __shfl_xor(s2, 16, 32);
  const float mu = s * (1.f / 128.f);
  const float rs = rsqrtf(s2 * (1.f / 128.f) - mu * mu + 1e-5f);
#pragma unroll
  for (int i = 0; i < 64; ++i) {
    const int c = half * 64 + i;
    sx[t * DIMC + c] = (_Float16)((loc[i] - mu) * rs * g[c] + bt[c]);
  }
}

// Block-cooperative LayerNorm of 64 tokens x 128ch (LDS f32 src -> LDS f16 dst).
// 4 threads per token; optional 32-wide kpos add (broadcast over heads).
__device__ __forceinline__ void block_ln64(const float* __restrict__ Xs,
                                           _Float16* __restrict__ out,
                                           const float* __restrict__ g,
                                           const float* __restrict__ bt,
                                           const float* __restrict__ pos32, int tid) {
  const int t = tid >> 2, q = tid & 3;
  const float* xr = Xs + t * DIMC + q * 32;
  float loc[32];
  float s = 0.f, s2 = 0.f;
#pragma unroll
  for (int i = 0; i < 32; ++i) {
    float x = xr[i];
    if (pos32) x += pos32[t * 32 + i];
    loc[i] = x; s += x; s2 += x * x;
  }
  s  += __shfl_xor(s, 1, 32);  s  += __shfl_xor(s, 2, 32);
  s2 += __shfl_xor(s2, 1, 32); s2 += __shfl_xor(s2, 2, 32);
  const float mu = s * (1.f / 128.f);
  const float rs = rsqrtf(s2 * (1.f / 128.f) - mu * mu + 1e-5f);
#pragma unroll
  for (int i = 0; i < 32; ++i) {
    const int c = q * 32 + i;
    out[t * DIMC + c] = (_Float16)((loc[i] - mu) * rs * g[c] + bt[c]);
  }
}

// ---------------------------------------------------------------------------
// Weight prep
// ---------------------------------------------------------------------------
__global__ void cvt_f16_kernel(const float* __restrict__ s, _Float16* __restrict__ d, int n) {
  const int i = blockIdx.x * 256 + threadIdx.x;
  if (i < n) d[i] = (_Float16)s[i];
}

// Fold per-stream LN affine into a [128,128] weight: W'[n,k]=W[n,k]*g[k]; b'[n]=sum_k b[k]*W[n,k]
__global__ void fold_ln_w_kernel(const float* __restrict__ W, const float* __restrict__ g,
                                 const float* __restrict__ b, _Float16* __restrict__ Wh,
                                 float* __restrict__ bout) {
  const int n = threadIdx.x;
  float s = 0.f;
  for (int k = 0; k < DIMC; ++k) {
    const float w = W[n * DIMC + k];
    Wh[n * DIMC + k] = (_Float16)(w * g[k]);
    s += b[k] * w;
  }
  bout[n] = s;
}

// ---------------------------------------------------------------------------
// Fused LN(+pos) -> X@W^T -> f16 out ([T,128] or, if TRANSPOSED, per-batch
// [B][128][TB_ENC]).  grid = T/128 blocks, 256 threads (8 waves x 16-token tiles)
// ---------------------------------------------------------------------------
template <bool TRANSPOSED>
__global__ __launch_bounds__(256)
void ln_gemm_kernel(const float* __restrict__ X, const float* __restrict__ pos,
                    const float* __restrict__ g, const float* __restrict__ bt,
                    const _Float16* __restrict__ W, _Float16* __restrict__ Y) {
  __shared__ _Float16 sx[8][16 * DIMC];
  const int wave = threadIdx.x >> 5, lane = threadIdx.x & 31;
  const int t0 = (blockIdx.x * 8 + wave) * 16;
  wave_ln16(X + (size_t)t0 * DIMC, pos ? pos + (size_t)t0 * DIMC : nullptr, g, bt, &sx[wave][0], lane);
  __syncthreads();
  v16h a[4];
#pragma unroll
  for (int kc = 0; kc < 4; ++kc) a[kc] = load_a(&sx[wave][0] + kc * 32, DIMC, lane);
  const int col = lane & 15, hb8 = (lane >> 4) * 8;
#pragma unroll
  for (int nt = 0; nt < 8; ++nt) {
    v8f acc = vzero8();
#pragma unroll
    for (int kc = 0; kc < 4; ++kc)
      acc = wmma16(a[kc], load_b(W + nt * 16 * DIMC + kc * 32, DIMC, lane), acc);
    if (!TRANSPOSED) {
#pragma unroll
      for (int r = 0; r < 8; ++r)
        Y[(size_t)(t0 + hb8 + r) * DIMC + nt * 16 + col] = (_Float16)acc[r];
    } else {
      const int bidx = t0 / TB_ENC, rloc = t0 - bidx * TB_ENC;  // compile-time divisor
      v8h pk;
#pragma unroll
      for (int r = 0; r < 8; ++r) pk[r] = (_Float16)acc[r];
      _Float16* yp = Y + (size_t)bidx * DIMC * TB_ENC + (size_t)(nt * 16 + col) * TB_ENC + rloc + hb8;
      *(v8h*)yp = pk;  // contiguous 16B store
    }
  }
}

// ---------------------------------------------------------------------------
// Encoder flash attention (1 head, hd=128): q[64,128] x K[3136,128], fused proj+residual
// grid = 32 batches, 128 threads (4 waves, one 16-query tile each)
// ---------------------------------------------------------------------------
__global__ __launch_bounds__(128)
void enc_flash_kernel(const _Float16* __restrict__ qe, const _Float16* __restrict__ ke,
                      const _Float16* __restrict__ vet, const _Float16* __restrict__ Wp,
                      const float* __restrict__ bp, const float* __restrict__ parts,
                      float* __restrict__ pr0) {
  __shared__ _Float16 Pst[4][16 * 32];
  __shared__ _Float16 Ost[4][16 * DIMC];
  const int bb = blockIdx.x;
  const int wave = threadIdx.x >> 5, lane = threadIdx.x & 31;
  const int m0 = wave * 16;
  const int col = lane & 15, hb8 = (lane >> 4) * 8;

  v16h aq[4];
  const _Float16* qp = qe + (size_t)(bb * 64 + m0) * DIMC;
#pragma unroll
  for (int kc = 0; kc < 4; ++kc) aq[kc] = load_a(qp + kc * 32, DIMC, lane);

  v8f o[8];
#pragma unroll
  for (int j = 0; j < 8; ++j) o[j] = vzero8();
  float mrow[8], lrow[8];
#pragma unroll
  for (int r = 0; r < 8; ++r) { mrow[r] = -1e30f; lrow[r] = 0.f; }

  const _Float16* kb = ke  + (size_t)bb * TB_ENC * DIMC;
  const _Float16* vb = vet + (size_t)bb * DIMC * TB_ENC;
  const float scale = 0.0883883476f;  // 1/sqrt(128)

  for (int nk = 0; nk < TB_ENC; nk += 32) {
    v8f s0 = vzero8(), s1 = vzero8();
#pragma unroll
    for (int kc = 0; kc < 4; ++kc) {
      s0 = wmma16(aq[kc], load_b(kb + (size_t)nk * DIMC + kc * 32, DIMC, lane), s0);
      s1 = wmma16(aq[kc], load_b(kb + (size_t)(nk + 16) * DIMC + kc * 32, DIMC, lane), s1);
    }
#pragma unroll
    for (int r = 0; r < 8; ++r) {
      const float v0 = s0[r] * scale, v1 = s1[r] * scale;
      const float mx = redmax16(fmaxf(v0, v1));
      const float mnew = fmaxf(mrow[r], mx);
      const float alpha = __expf(mrow[r] - mnew);
      const float e0 = __expf(v0 - mnew), e1 = __expf(v1 - mnew);
      lrow[r] = lrow[r] * alpha + redsum16(e0 + e1);
      mrow[r] = mnew;
#pragma unroll
      for (int j = 0; j < 8; ++j) o[j][r] *= alpha;
      Pst[wave][(hb8 + r) * 32 + col]      = (_Float16)e0;
      Pst[wave][(hb8 + r) * 32 + 16 + col] = (_Float16)e1;
    }
    __syncthreads();
    const v16h ap = load_a(&Pst[wave][0], 32, lane);
#pragma unroll
    for (int j = 0; j < 8; ++j)
      o[j] = wmma16(ap, load_b(vb + (size_t)j * 16 * TB_ENC + nk, TB_ENC, lane), o[j]);
    __syncthreads();
  }
#pragma unroll
  for (int j = 0; j < 8; ++j)
#pragma unroll
    for (int r = 0; r < 8; ++r)
      Ost[wave][(hb8 + r) * DIMC + j * 16 + col] = (_Float16)(o[j][r] / lrow[r]);
  __syncthreads();
  v16h ao[4];
#pragma unroll
  for (int kc = 0; kc < 4; ++kc) ao[kc] = load_a(&Ost[wave][0] + kc * 32, DIMC, lane);
  const float* pres = parts + (size_t)(bb * 64 + m0) * DIMC;
#pragma unroll
  for (int nt = 0; nt < 8; ++nt) {
    v8f acc = vzero8();
#pragma unroll
    for (int kc = 0; kc < 4; ++kc)
      acc = wmma16(ao[kc], load_b(Wp + nt * 16 * DIMC + kc * 32, DIMC, lane), acc);
#pragma unroll
    for (int r = 0; r < 8; ++r)
      pr0[(size_t)(bb * 64 + m0 + hb8 + r) * DIMC + nt * 16 + col] =
          acc[r] + bp[nt * 16 + col] + pres[(hb8 + r) * DIMC + nt * 16 + col];
  }
}

// ---------------------------------------------------------------------------
// Parts update: reason mix + encoder FFN + decoder K/V projection.
// grid = 32 batches, 256 threads, 64 KB dynamic LDS.
// ---------------------------------------------------------------------------
__global__ __launch_bounds__(256)
void parts_update_kernel(const float* __restrict__ pr0, const _Float16* __restrict__ reasonW,
                         const float* __restrict__ rg, const float* __restrict__ rb,
                         const float* __restrict__ eg, const float* __restrict__ eb,
                         const _Float16* __restrict__ ew1, const float* __restrict__ eb1,
                         const _Float16* __restrict__ ew2, const float* __restrict__ eb2,
                         const float* __restrict__ kg, const float* __restrict__ kbn,
                         const _Float16* __restrict__ wk1,
                         const float* __restrict__ vg, const float* __restrict__ vbn,
                         const _Float16* __restrict__ wv1,
                         const float* __restrict__ kpos,
                         float* __restrict__ pr_out, _Float16* __restrict__ deck,
                         _Float16* __restrict__ decvt) {
  extern __shared__ char smem[];
  float*     prb = (float*)smem;                       // 64*128*4 = 32768
  _Float16*  xln = (_Float16*)(smem + 32768);          // 64*128*2 = 16384
  _Float16*  hbf = (_Float16*)(smem + 49152);          // 16384
  const int bb = blockIdx.x;
  const int tid = threadIdx.x, wave = tid >> 5, lane = tid & 31;
  const int col = lane & 15, hb8 = (lane >> 4) * 8;

  for (int i = tid; i < 64 * DIMC; i += 256) prb[i] = pr0[(size_t)bb * 8192 + i];
  __syncthreads();

  // --- reasoning: pr += reason_w @ LN(pr) ---
  block_ln64(prb, xln, rg, rb, nullptr, tid);
  __syncthreads();
  for (int tix = wave; tix < 32; tix += 8) {
    const int mt = tix >> 3, nt = tix & 7;
    v8f acc = vzero8();
#pragma unroll
    for (int kc = 0; kc < 2; ++kc)
      acc = wmma16(load_a(reasonW + mt * 16 * 64 + kc * 32, 64, lane),
                   load_bT(xln + kc * 32 * DIMC + nt * 16, DIMC, lane), acc);
#pragma unroll
    for (int r = 0; r < 8; ++r)
      prb[(mt * 16 + hb8 + r) * DIMC + nt * 16 + col] += acc[r];
  }
  __syncthreads();

  // --- encoder FFN (hidden = 128) ---
  block_ln64(prb, xln, eg, eb, nullptr, tid);
  __syncthreads();
  for (int tix = wave; tix < 32; tix += 8) {
    const int mt = tix >> 3, nt = tix & 7;
    v8f acc = vzero8();
#pragma unroll
    for (int kc = 0; kc < 4; ++kc)
      acc = wmma16(load_a(xln + mt * 16 * DIMC + kc * 32, DIMC, lane),
                   load_b(ew1 + nt * 16 * DIMC + kc * 32, DIMC, lane), acc);
    const float bbi = eb1[nt * 16 + col];
#pragma unroll
    for (int r = 0; r < 8; ++r)
      hbf[(mt * 16 + hb8 + r) * DIMC + nt * 16 + col] = (_Float16)gelu_exact(acc[r] + bbi);
  }
  __syncthreads();
  for (int tix = wave; tix < 32; tix += 8) {
    const int mt = tix >> 3, nt = tix & 7;
    v8f acc = vzero8();
#pragma unroll
    for (int kc = 0; kc < 4; ++kc)
      acc = wmma16(load_a(hbf + mt * 16 * DIMC + kc * 32, DIMC, lane),
                   load_b(ew2 + nt * 16 * DIMC + kc * 32, DIMC, lane), acc);
    const float bbi = eb2[nt * 16 + col];
#pragma unroll
    for (int r = 0; r < 8; ++r)
      prb[(mt * 16 + hb8 + r) * DIMC + nt * 16 + col] += acc[r] + bbi;
  }
  __syncthreads();

  for (int i = tid; i < 64 * DIMC; i += 256) pr_out[(size_t)bb * 8192 + i] = prb[i];

  // --- decoder K = LN(pr + kpos) @ Wk^T ---
  block_ln64(prb, xln, kg, kbn, kpos + (size_t)bb * 64 * 32, tid);
  __syncthreads();
  for (int tix = wave; tix < 32; tix += 8) {
    const int mt = tix >> 3, nt = tix & 7;
    v8f acc = vzero8();
#pragma unroll
    for (int kc = 0; kc < 4; ++kc)
      acc = wmma16(load_a(xln + mt * 16 * DIMC + kc * 32, DIMC, lane),
                   load_b(wk1 + nt * 16 * DIMC + kc * 32, DIMC, lane), acc);
#pragma unroll
    for (int r = 0; r < 8; ++r)
      deck[(size_t)(bb * 64 + mt * 16 + hb8 + r) * DIMC + nt * 16 + col] = (_Float16)acc[r];
  }
  __syncthreads();

  // --- decoder V = LN(pr) @ Wv^T (stored transposed [128][64]) ---
  block_ln64(prb, xln, vg, vbn, nullptr, tid);
  __syncthreads();
  for (int tix = wave; tix < 32; tix += 8) {
    const int mt = tix >> 3, nt = tix & 7;
    v8f acc = vzero8();
#pragma unroll
    for (int kc = 0; kc < 4; ++kc)
      acc = wmma16(load_a(xln + mt * 16 * DIMC + kc * 32, DIMC, lane),
                   load_b(wv1 + nt * 16 * DIMC + kc * 32, DIMC, lane), acc);
    v8h pk;
#pragma unroll
    for (int r = 0; r < 8; ++r) pk[r] = (_Float16)acc[r];
    *(v8h*)(decvt + (size_t)bb * 8192 + (size_t)(nt * 16 + col) * 64 + mt * 16 + hb8) = pk;
  }
}

// ---------------------------------------------------------------------------
// Decoder cross attention (4 heads, hd=32, 64 keys) fused proj+residual.
// grid = 784 blocks x 256 threads; each wave owns a 16-query tile.
// ---------------------------------------------------------------------------
__global__ __launch_bounds__(256)
void dec_cross_kernel(const _Float16* __restrict__ qd, const _Float16* __restrict__ deck,
                      const _Float16* __restrict__ decvt, const _Float16* __restrict__ Wp,
                      const float* __restrict__ bp, const float* __restrict__ x,
                      float* __restrict__ out) {
  __shared__ _Float16 Pst[8][16 * 64];
  __shared__ _Float16 Ost[8][16 * DIMC];
  const int wave = threadIdx.x >> 5, lane = threadIdx.x & 31;
  const int t0 = (blockIdx.x * 8 + wave) * 16;
  const int bb = t0 / TB_ENC;
  const int col = lane & 15, hb8 = (lane >> 4) * 8;
  const _Float16* kb = deck  + (size_t)bb * 64 * DIMC;
  const _Float16* vb = decvt + (size_t)bb * 8192;
  const float scale = 0.1767766953f;  // 1/sqrt(32)

  v8f oo[8];
#pragma unroll
  for (int j = 0; j < 8; ++j) oo[j] = vzero8();

  for (int h = 0; h < 4; ++h) {
    const v16h aq = load_a(qd + (size_t)t0 * DIMC + h * 32, DIMC, lane);
    v8f s[4];
#pragma unroll
    for (int j = 0; j < 4; ++j)
      s[j] = wmma16(aq, load_b(kb + j * 16 * DIMC + h * 32, DIMC, lane), vzero8());
#pragma unroll
    for (int r = 0; r < 8; ++r) {
      float v0 = s[0][r] * scale, v1 = s[1][r] * scale, v2 = s[2][r] * scale, v3 = s[3][r] * scale;
      const float mx = redmax16(fmaxf(fmaxf(v0, v1), fmaxf(v2, v3)));
      v0 = __expf(v0 - mx); v1 = __expf(v1 - mx); v2 = __expf(v2 - mx); v3 = __expf(v3 - mx);
      const float inv = 1.f / redsum16(v0 + v1 + v2 + v3);
      Pst[wave][(hb8 + r) * 64 + 0 * 16 + col] = (_Float16)(v0 * inv);
      Pst[wave][(hb8 + r) * 64 + 1 * 16 + col] = (_Float16)(v1 * inv);
      Pst[wave][(hb8 + r) * 64 + 2 * 16 + col] = (_Float16)(v2 * inv);
      Pst[wave][(hb8 + r) * 64 + 3 * 16 + col] = (_Float16)(v3 * inv);
    }
    __syncthreads();
#pragma unroll
    for (int cc = 0; cc < 2; ++cc) {
      v8f acc = oo[h * 2 + cc];
#pragma unroll
      for (int kc2 = 0; kc2 < 2; ++kc2)
        acc = wmma16(load_a(&Pst[wave][0] + kc2 * 32, 64, lane),
                     load_b(vb + (h * 32 + cc * 16) * 64 + kc2 * 32, 64, lane), acc);
      oo[h * 2 + cc] = acc;
    }
    __syncthreads();
  }
#pragma unroll
  for (int j = 0; j < 8; ++j)
#pragma unroll
    for (int r = 0; r < 8; ++r)
      Ost[wave][(hb8 + r) * DIMC + j * 16 + col] = (_Float16)oo[j][r];
  __syncthreads();
  v16h ao[4];
#pragma unroll
  for (int kc = 0; kc < 4; ++kc) ao[kc] = load_a(&Ost[wave][0] + kc * 32, DIMC, lane);
#pragma unroll
  for (int nt = 0; nt < 8; ++nt) {
    v8f acc = vzero8();
#pragma unroll
    for (int kc = 0; kc < 4; ++kc)
      acc = wmma16(ao[kc], load_b(Wp + nt * 16 * DIMC + kc * 32, DIMC, lane), acc);
#pragma unroll
    for (int r = 0; r < 8; ++r) {
      const size_t idx = (size_t)(t0 + hb8 + r) * DIMC + nt * 16 + col;
      out[idx] = acc[r] + bp[nt * 16 + col] + x[idx];
    }
  }
}

// ---------------------------------------------------------------------------
// Fused FFN: Y = X + GELU(LN(X)@W1^T + b1)@W2^T + b2.  Hidden chunked by 32,
// never spilled to HBM.  grid = 784 x 256.  H is compile-time (512).
// ---------------------------------------------------------------------------
template <int H>
__global__ __launch_bounds__(256)
void ffn_kernel(const float* __restrict__ X, const float* __restrict__ g,
                const float* __restrict__ bt, const _Float16* __restrict__ W1,
                const float* __restrict__ b1, const _Float16* __restrict__ W2,
                const float* __restrict__ b2, float* __restrict__ out) {
  __shared__ _Float16 sx[8][16 * DIMC];
  __shared__ _Float16 sh[8][16 * 32];
  const int wave = threadIdx.x >> 5, lane = threadIdx.x & 31;
  const int t0 = (blockIdx.x * 8 + wave) * 16;
  const int col = lane & 15, hb8 = (lane >> 4) * 8;

  wave_ln16(X + (size_t)t0 * DIMC, nullptr, g, bt, &sx[wave][0], lane);
  __syncthreads();
  v16h ax[4];
#pragma unroll
  for (int kc = 0; kc < 4; ++kc) ax[kc] = load_a(&sx[wave][0] + kc * 32, DIMC, lane);

  v8f y[8];
#pragma unroll
  for (int j = 0; j < 8; ++j) y[j] = vzero8();

  for (int hc = 0; hc < H; hc += 32) {
    v8f h0 = vzero8(), h1 = vzero8();
#pragma unroll
    for (int kc = 0; kc < 4; ++kc) {
      h0 = wmma16(ax[kc], load_b(W1 + (size_t)hc * DIMC + kc * 32, DIMC, lane), h0);
      h1 = wmma16(ax[kc], load_b(W1 + (size_t)(hc + 16) * DIMC + kc * 32, DIMC, lane), h1);
    }
    const float bb0 = b1[hc + col], bb1 = b1[hc + 16 + col];
#pragma unroll
    for (int r = 0; r < 8; ++r) {
      sh[wave][(hb8 + r) * 32 + col]      = (_Float16)gelu_exact(h0[r] + bb0);
      sh[wave][(hb8 + r) * 32 + 16 + col] = (_Float16)gelu_exact(h1[r] + bb1);
    }
    __syncthreads();
    const v16h ah = load_a(&sh[wave][0], 32, lane);
#pragma unroll
    for (int j = 0; j < 8; ++j)
      y[j] = wmma16(ah, load_b(W2 + (size_t)(j * 16) * H + hc, H, lane), y[j]);
    __syncthreads();
  }
#pragma unroll
  for (int j = 0; j < 8; ++j)
#pragma unroll
    for (int r = 0; r < 8; ++r) {
      const size_t idx = (size_t)(t0 + hb8 + r) * DIMC + j * 16 + col;
      out[idx] = y[j][r] + b2[j * 16 + col] + X[idx];
    }
}

// ---------------------------------------------------------------------------
// Local window self-attention (49 tokens padded to 64, 4 heads, rel-pos bias).
// LN affine folded into Wq/Wk/Wv (bq/bk/bv). grid = 2048 windows x 256 threads,
// 88 KB dynamic LDS.
// ---------------------------------------------------------------------------
__global__ __launch_bounds__(256)
void local_attn_kernel(const float* __restrict__ X,
                       const _Float16* __restrict__ Wq, const _Float16* __restrict__ Wk,
                       const _Float16* __restrict__ Wv, const float* __restrict__ bq,
                       const float* __restrict__ bk, const float* __restrict__ bv,
                       const _Float16* __restrict__ Wp, const float* __restrict__ bp,
                       const float* __restrict__ rel_h, const float* __restrict__ rel_w,
                       float* __restrict__ out) {
  extern __shared__ char smem[];
  _Float16* Xst = (_Float16*)smem;                 // 16384 (reused as Ofull)
  _Float16* Kst = (_Float16*)(smem + 16384);       // 16384
  _Float16* Vt  = (_Float16*)(smem + 32768);       // 16384
  _Float16* Qsm = (_Float16*)(smem + 49152);       // 8 waves * 2 combos * 16*32 = 16384
  _Float16* Pst = (_Float16*)(smem + 65536);       // 8 waves * 16*64 = 16384
  float*    blb = (float*)(smem + 81920);          // 8*16*8 = 4096
  float*    bwb = (float*)(smem + 86016);          // 4096
  _Float16* Ofull = Xst;

  const int win = blockIdx.x;
  const int tid = threadIdx.x, wave = tid >> 5, lane = tid & 31;
  const int col = lane & 15, hb8 = (lane >> 4) * 8;
  const float* xw = X + (size_t)win * KW * DIMC;

  // parameter-free LN of the window into Xst (rows >= 49 zero-padded)
  {
    const int t = tid >> 2, q = tid & 3;
    if (t < KW) {
      const float* xr = xw + t * DIMC + q * 32;
      float loc[32];
      float s = 0.f, s2 = 0.f;
#pragma unroll
      for (int i = 0; i < 32; ++i) { const float v = xr[i]; loc[i] = v; s += v; s2 += v * v; }
      s  += __shfl_xor(s, 1, 32);  s  += __shfl_xor(s, 2, 32);
      s2 += __shfl_xor(s2, 1, 32); s2 += __shfl_xor(s2, 2, 32);
      const float mu = s * (1.f / 128.f);
      const float rs = rsqrtf(s2 * (1.f / 128.f) - mu * mu + 1e-5f);
#pragma unroll
      for (int i = 0; i < 32; ++i) Xst[t * DIMC + q * 32 + i] = (_Float16)((loc[i] - mu) * rs);
    } else {
#pragma unroll
      for (int i = 0; i < 32; ++i) Xst[t * DIMC + q * 32 + i] = (_Float16)0.f;
    }
  }
  __syncthreads();

  // Q slabs: combo c = h*4 + m; wave owns c = wave and wave+8
#pragma unroll
  for (int ci = 0; ci < 2; ++ci) {
    const int c = wave + ci * 8, h = c >> 2, m = c & 3;
#pragma unroll
    for (int nt2 = 0; nt2 < 2; ++nt2) {
      const int n0 = h * 32 + nt2 * 16;
      v8f acc = vzero8();
#pragma unroll
      for (int kc = 0; kc < 4; ++kc)
        acc = wmma16(load_a(Xst + m * 16 * DIMC + kc * 32, DIMC, lane),
                     load_b(Wq + n0 * DIMC + kc * 32, DIMC, lane), acc);
      const float bbi = bq[n0 + col];
#pragma unroll
      for (int r = 0; r < 8; ++r)
        Qsm[wave * 1024 + ci * 512 + (hb8 + r) * 32 + nt2 * 16 + col] = (_Float16)(acc[r] + bbi);
    }
  }
  // K and V: 64 tiles across 8 waves
  for (int tix = wave; tix < 64; tix += 8) {
    const int isv = tix >> 5, mt = (tix >> 3) & 3, nt = tix & 7;
    const _Float16* Wm = isv ? Wv : Wk;
    v8f acc = vzero8();
#pragma unroll
    for (int kc = 0; kc < 4; ++kc)
      acc = wmma16(load_a(Xst + mt * 16 * DIMC + kc * 32, DIMC, lane),
                   load_b(Wm + nt * 16 * DIMC + kc * 32, DIMC, lane), acc);
    const float bbi = (isv ? bv : bk)[nt * 16 + col];
    if (!isv) {
#pragma unroll
      for (int r = 0; r < 8; ++r)
        Kst[(mt * 16 + hb8 + r) * DIMC + nt * 16 + col] = (_Float16)(acc[r] + bbi);
    } else {
#pragma unroll
      for (int r = 0; r < 8; ++r)
        Vt[(nt * 16 + col) * 64 + mt * 16 + hb8 + r] = (_Float16)(acc[r] + bbi);
    }
  }
  __syncthreads();  // Xst free below (becomes Ofull)

  const float scale = 0.1767766953f;  // 1/sqrt(32)
#pragma unroll
  for (int ci = 0; ci < 2; ++ci) {
    const int c = wave + ci * 8, h = c >> 2, m = c & 3;
    const _Float16* Qsl = Qsm + wave * 1024 + ci * 512;
    // rel-pos bias rows (lane<16 -> lh using first 16 ch; lane>=16 -> lw using last 16)
    {
      const int rloc = lane & 15;
      const int tok = m * 16 + rloc;
      const int xq = (tok < KW) ? tok / 7 : 0;
      const int yq = (tok < KW) ? tok % 7 : 0;
      const float* relT = (lane < 16) ? rel_h : rel_w;
      const int xy = (lane < 16) ? xq : yq;
      const int coff = (lane < 16) ? 0 : 16;
      float* dst = ((lane < 16) ? blb : bwb) + wave * 128 + rloc * 8;
#pragma unroll
      for (int kx = 0; kx < 7; ++kx) {
        const float* rv = relT + (kx - xy + 6) * 16;
        float sdot = 0.f;
#pragma unroll
        for (int c2 = 0; c2 < 16; ++c2) sdot += (float)Qsl[rloc * 32 + coff + c2] * rv[c2];
        dst[kx] = sdot;
      }
    }
    __syncthreads();
    const v16h aqf = load_a(Qsl, 32, lane);
    v8f sv[4];
#pragma unroll
    for (int j = 0; j < 4; ++j)
      sv[j] = wmma16(aqf, load_b(Kst + j * 16 * DIMC + h * 32, DIMC, lane), vzero8());
#pragma unroll
    for (int r = 0; r < 8; ++r) {
      float vv[4];
#pragma unroll
      for (int j = 0; j < 4; ++j) {
        const int key = j * 16 + col;
        if (key < KW) {
          const float bias = blb[wave * 128 + (hb8 + r) * 8 + key / 7] +
                             bwb[wave * 128 + (hb8 + r) * 8 + key % 7];
          vv[j] = (sv[j][r] + bias) * scale;
        } else {
          vv[j] = -1e30f;
        }
      }
      const float mx = redmax16(fmaxf(fmaxf(vv[0], vv[1]), fmaxf(vv[2], vv[3])));
#pragma unroll
      for (int j = 0; j < 4; ++j) vv[j] = __expf(vv[j] - mx);
      const float inv = 1.f / redsum16(vv[0] + vv[1] + vv[2] + vv[3]);
#pragma unroll
      for (int j = 0; j < 4; ++j)
        Pst[wave * 1024 + (hb8 + r) * 64 + j * 16 + col] = (_Float16)(vv[j] * inv);
    }
    __syncthreads();
#pragma unroll
    for (int cc = 0; cc < 2; ++cc) {
      v8f acc = vzero8();
#pragma unroll
      for (int kc2 = 0; kc2 < 2; ++kc2)
        acc = wmma16(load_a(Pst + wave * 1024 + kc2 * 32, 64, lane),
                     load_b(Vt + (h * 32 + cc * 16) * 64 + kc2 * 32, 64, lane), acc);
#pragma unroll
      for (int r = 0; r < 8; ++r)
        Ofull[(m * 16 + hb8 + r) * DIMC + h * 32 + cc * 16 + col] = (_Float16)acc[r];
    }
    __syncthreads();
  }

  // output projection + residual (only the 49 real tokens)
  for (int tix = wave; tix < 32; tix += 8) {
    const int mt = tix >> 3, nt = tix & 7;
    v8f acc = vzero8();
#pragma unroll
    for (int kc = 0; kc < 4; ++kc)
      acc = wmma16(load_a(Ofull + mt * 16 * DIMC + kc * 32, DIMC, lane),
                   load_b(Wp + nt * 16 * DIMC + kc * 32, DIMC, lane), acc);
#pragma unroll
    for (int r = 0; r < 8; ++r) {
      const int tok = mt * 16 + hb8 + r;
      if (tok < KW)
        out[(size_t)win * KW * DIMC + tok * DIMC + nt * 16 + col] =
            acc[r] + bp[nt * 16 + col] + xw[tok * DIMC + nt * 16 + col];
    }
  }
}

// ---------------------------------------------------------------------------
// Host-side orchestration
// ---------------------------------------------------------------------------
extern "C" void kernel_launch(void* const* d_in, const int* in_sizes, int n_in,
                              void* d_out, int out_size, void* d_ws, size_t ws_size,
                              hipStream_t stream) {
  (void)in_sizes; (void)n_in; (void)out_size; (void)ws_size;
  const float* x     = (const float*)d_in[0];
  const float* parts = (const float*)d_in[1];
  const float* qpos  = (const float*)d_in[2];
  const float* kpos  = (const float*)d_in[3];
  const float* lng   = (const float*)d_in[4];
  const float* lnb   = (const float*)d_in[5];
  const float* qkvw  = (const float*)d_in[6];
  const float* projw = (const float*)d_in[7];
  const float* projb = (const float*)d_in[8];
  const float* rlg   = (const float*)d_in[9];
  const float* rlb   = (const float*)d_in[10];
  const float* reasw = (const float*)d_in[11];
  const float* eg    = (const float*)d_in[12];
  const float* eb    = (const float*)d_in[13];
  const float* ew1   = (const float*)d_in[14];
  const float* eb1   = (const float*)d_in[15];
  const float* ew2   = (const float*)d_in[16];
  const float* eb2   = (const float*)d_in[17];
  const float* f1g   = (const float*)d_in[18];
  const float* f1b   = (const float*)d_in[19];
  const float* f1w1  = (const float*)d_in[20];
  const float* f1b1  = (const float*)d_in[21];
  const float* f1w2  = (const float*)d_in[22];
  const float* f1b2  = (const float*)d_in[23];
  const float* f2g   = (const float*)d_in[24];
  const float* f2b   = (const float*)d_in[25];
  const float* f2w1  = (const float*)d_in[26];
  const float* f2b1  = (const float*)d_in[27];
  const float* f2w2  = (const float*)d_in[28];
  const float* f2b2  = (const float*)d_in[29];
  const float* relh  = (const float*)d_in[30];
  const float* relw  = (const float*)d_in[31];

  char* ws = (char*)d_ws;
  size_t off = 0;
  auto carve = [&](size_t bytes) -> char* {
    char* p = ws + off;
    off += (bytes + 255) & ~(size_t)255;
    return p;
  };
  _Float16* qkvh  = (_Float16*)carve((size_t)9 * 16384 * 2);
  _Float16* projh = (_Float16*)carve((size_t)3 * 16384 * 2);
  _Float16* reash = (_Float16*)carve(4096 * 2);
  _Float16* ew1h  = (_Float16*)carve(16384 * 2);
  _Float16* ew2h  = (_Float16*)carve(16384 * 2);
  _Float16* f1w1h = (_Float16*)carve(65536 * 2);
  _Float16* f1w2h = (_Float16*)carve(65536 * 2);
  _Float16* f2w1h = (_Float16*)carve(65536 * 2);
  _Float16* f2w2h = (_Float16*)carve(65536 * 2);
  float*    lqb   = (float*)carve(3 * 128 * 4);        // folded local-attn ln biases
  _Float16* qe    = (_Float16*)carve((size_t)2048 * 128 * 2);
  _Float16* ke    = (_Float16*)carve((size_t)T_TOK * 128 * 2);
  _Float16* vet   = (_Float16*)carve((size_t)T_TOK * 128 * 2);
  _Float16* qd    = (_Float16*)carve((size_t)T_TOK * 128 * 2);
  _Float16* deck  = (_Float16*)carve((size_t)2048 * 128 * 2);
  _Float16* decvt = (_Float16*)carve((size_t)2048 * 128 * 2);
  float*    pr0   = (float*)carve((size_t)2048 * 128 * 4);
  float*    buf1  = (float*)carve((size_t)T_TOK * 128 * 4);

  float* feats_out = (float*)d_out;
  float* pr_out    = feats_out + (size_t)T_TOK * 128;

  auto cvt = [&](const float* s, _Float16* d, int n) {
    cvt_f16_kernel<<<(n + 255) / 256, 256, 0, stream>>>(s, d, n);
  };
  // plain conversions: enc (3) + dec-cross (3) qkv weights; local (3) are LN-folded
  cvt(qkvw, qkvh, 6 * 16384);
  for (int j = 0; j < 3; ++j)
    fold_ln_w_kernel<<<1, 128, 0, stream>>>(qkvw + (size_t)(6 + j) * 16384,
                                            lng + (6 + j) * 128, lnb + (6 + j) * 128,
                                            qkvh + (size_t)(6 + j) * 16384, lqb + j * 128);
  cvt(projw, projh, 3 * 16384);
  cvt(reasw, reash, 4096);
  cvt(ew1, ew1h, 16384);
  cvt(ew2, ew2h, 16384);
  cvt(f1w1, f1w1h, 65536);
  cvt(f1w2, f1w2h, 65536);
  cvt(f2w1, f2w1h, 65536);
  cvt(f2w2, f2w2h, 65536);

  // ---- Encoder: q/k/v projections then flash attention + proj + residual ----
  ln_gemm_kernel<false><<<16, 256, 0, stream>>>(parts, qpos, lng + 0, lnb + 0, qkvh + 0, qe);
  ln_gemm_kernel<false><<<784, 256, 0, stream>>>(x, nullptr, lng + 128, lnb + 128,
                                                 qkvh + 16384, ke);
  ln_gemm_kernel<true><<<784, 256, 0, stream>>>(x, nullptr, lng + 256, lnb + 256,
                                                qkvh + 2 * 16384, vet);
  enc_flash_kernel<<<32, 128, 0, stream>>>(qe, ke, vet, projh, projb, parts, pr0);

  // ---- Parts: reasoning + encoder FFN + decoder K/V; pr -> d_out tail ----
  parts_update_kernel<<<32, 256, 65536, stream>>>(
      pr0, reash, rlg, rlb, eg, eb, ew1h, eb1, ew2h, eb2,
      lng + 4 * 128, lnb + 4 * 128, qkvh + (size_t)4 * 16384,
      lng + 5 * 128, lnb + 5 * 128, qkvh + (size_t)5 * 16384,
      kpos, pr_out, deck, decvt);

  // ---- Decoder cross-attention ----
  ln_gemm_kernel<false><<<784, 256, 0, stream>>>(x, nullptr, lng + 3 * 128, lnb + 3 * 128,
                                                 qkvh + (size_t)3 * 16384, qd);
  dec_cross_kernel<<<784, 256, 0, stream>>>(qd, deck, decvt, projh + 16384, projb + 128,
                                            x, buf1);

  // ---- FFN1 (in-place) ----
  ffn_kernel<512><<<784, 256, 0, stream>>>(buf1, f1g, f1b, f1w1h, f1b1, f1w2h, f1b2, buf1);

  // ---- Local window attention with rel-pos bias (in-place) ----
  local_attn_kernel<<<NWIN, 256, 90112, stream>>>(
      buf1, qkvh + (size_t)6 * 16384, qkvh + (size_t)7 * 16384, qkvh + (size_t)8 * 16384,
      lqb, lqb + 128, lqb + 256, projh + 2 * 16384, projb + 256, relh, relw, buf1);

  // ---- FFN2 -> feats output ----
  ffn_kernel<512><<<784, 256, 0, stream>>>(buf1, f2g, f2b, f2w1h, f2b1, f2w2h, f2b2,
                                           feats_out);
}
